// EfficientAdditiveAttention_39032662786431
// MI455X (gfx1250) — matrix operand order repaired and checked by reading, not compiled, not run on hardware
//
#include <hip/hip_runtime.h>
#include <hip/hip_bf16.h>

// ---------------------------------------------------------------------------
// EfficientAdditiveAttention on MI455X (gfx1250), wave32 + WMMA bf16.
//   B=16, N=2048, IN_DIMS=D=TOKEN_DIM=768, M = B*N = 32768.
// Four [M,768]x[768,768] GEMMs done with v_wmma_f32_16x16x32_bf16.
// Wave tile 64x64 (16 WMMAs per 8 fragment loads), WG tile 128x256.
// ---------------------------------------------------------------------------

typedef __attribute__((ext_vector_type(16))) __bf16 bf16x16;
typedef __attribute__((ext_vector_type(8)))  __bf16 bf16x8;
typedef __attribute__((ext_vector_type(8)))  float  f32x8;

#define EPS 1e-12f

// ---------------------------------------------------------------------------
// Elementwise fp32 -> bf16 conversion (x -> xb), 4 elements/thread.
// ---------------------------------------------------------------------------
__global__ __launch_bounds__(256) void eaa_convert_x(const float* __restrict__ x,
                                                     __bf16* __restrict__ xb, int n) {
    int i = (blockIdx.x * 256 + threadIdx.x) * 4;
    if (i + 3 < n) {
        float4 v = *(const float4*)(x + i);
        xb[i + 0] = (__bf16)v.x;
        xb[i + 1] = (__bf16)v.y;
        xb[i + 2] = (__bf16)v.z;
        xb[i + 3] = (__bf16)v.w;
    } else {
        for (int j = i; j < n; ++j) xb[j] = (__bf16)x[j];
    }
}

// ---------------------------------------------------------------------------
// Weight transpose + convert: W [K,N] row-major fp32  ->  WT [N,K] bf16.
// Lets WMMA B-fragments load contiguous K per lane (lane = output column).
// ---------------------------------------------------------------------------
__global__ __launch_bounds__(256) void eaa_transpose_w(const float* __restrict__ W,
                                                       __bf16* __restrict__ WT,
                                                       int K, int N) {
    int idx = blockIdx.x * 256 + threadIdx.x;
    if (idx < N * K) {
        int n = idx / K;
        int k = idx - n * K;
        WT[idx] = (__bf16)W[(size_t)k * N + n];
    }
}

// ---------------------------------------------------------------------------
// bf16 WMMA GEMM:  out[M,N] = A[M,K](bf16) * BT[N,K]^T(bf16) + bias (+extra)
// WG = 256 threads = 8 waves arranged 2(M) x 4(N); WG tile 128 x 256.
// Wave tile 64(M) x 64(N): 4 A-frags x 4 B-frags -> 16 WMMAs / k-step of 32.
// MODE 0: f32 out = acc + bias
// MODE 1: bf16 out = acc + bias
// MODE 2: bf16 out = acc + bias + extra_f32[row,col]   (fused residual)
// ---------------------------------------------------------------------------
template <int MODE>
__global__ __launch_bounds__(256) void eaa_gemm_bf16(
    const __bf16* __restrict__ A,    // [M,K] row-major bf16
    const __bf16* __restrict__ BT,   // [N,K] row-major bf16 (B transposed)
    const float* __restrict__ bias,  // [N]
    const float* __restrict__ extra, // [M,N] f32 (MODE 2) or nullptr
    void* __restrict__ out,          // f32 (MODE 0) / bf16 (MODE 1,2) [M,N]
    int M, int N, int K) {
    const int lane    = threadIdx.x & 31;
    const int wave    = threadIdx.x >> 5;
    const int mWave   = wave & 1;   // 2 waves stacked in M
    const int nWave   = wave >> 1;  // 4 waves across N
    const int nBlocks = N / 256;
    const int bm      = blockIdx.x / nBlocks;
    const int bn      = blockIdx.x - bm * nBlocks;
    const int mBase   = bm * 128 + mWave * 64;
    const int nBase   = bn * 256 + nWave * 64;

    const int halfSel = lane >> 4;  // which 16-lane half of the wave
    const int l16     = lane & 15;

    // A fragment row offsets (same rows for both lane halves; halves differ in K).
    size_t rowOffA[4];
#pragma unroll
    for (int f = 0; f < 4; ++f) rowOffA[f] = (size_t)(mBase + f * 16 + l16) * K;

    f32x8 acc[4][4] = {};

    for (int k0 = 0; k0 < K; k0 += 32) {
        // ---- A fragments: 16-bit A 16x32 layout ----
        // lanes 0-15 : elems 0-7 = K k0..k0+7,  elems 8-15 = K k0+16..k0+23
        // lanes 16-31: elems 0-7 = K k0+8..+15, elems 8-15 = K k0+24..+31
        bf16x16 afr[4];
#pragma unroll
        for (int f = 0; f < 4; ++f) {
            const __bf16* p = A + rowOffA[f] + k0 + halfSel * 8;
            *(bf16x8*)&afr[f]       = *(const bf16x8*)p;
            *((bf16x8*)&afr[f] + 1) = *(const bf16x8*)(p + 16);
        }
        // ---- B fragments: 32x16 bf16, lane = column ----
        // lanes 0-15 : elems 0-7 = K k0..k0+7,  elems 8-15 = K k0+16..k0+23
        // lanes 16-31: elems 0-7 = K k0+8..+15, elems 8-15 = K k0+24..+31
        bf16x16 bfr[4];
#pragma unroll
        for (int t = 0; t < 4; ++t) {
            const __bf16* p = BT + (size_t)(nBase + t * 16 + l16) * K + k0 + halfSel * 8;
            *(bf16x8*)&bfr[t]       = *(const bf16x8*)p;
            *((bf16x8*)&bfr[t] + 1) = *(const bf16x8*)(p + 16);
        }

        // Unconditional speculative prefetch of next K-tile (global_prefetch_b8;
        // past-the-end prefetches are dropped silently per ISA).
        __builtin_prefetch(A + rowOffA[0] + k0 + 32, 0, 1);
        __builtin_prefetch(BT + (size_t)(nBase + l16) * K + k0 + 32, 0, 1);

#pragma unroll
        for (int i = 0; i < 4; ++i)
#pragma unroll
            for (int t = 0; t < 4; ++t)
                acc[i][t] = __builtin_amdgcn_wmma_f32_16x16x32_bf16(
                    false, afr[i], false, bfr[t], (short)0, acc[i][t], false, false);
    }

    // ---- Epilogue. C layout: VGPR r, lanes 0-15 -> M=r, lanes 16-31 -> M=8+r.
#pragma unroll
    for (int i = 0; i < 4; ++i) {
#pragma unroll
        for (int t = 0; t < 4; ++t) {
            const int col  = nBase + t * 16 + l16;
            const float bv = bias[col];
#pragma unroll
            for (int r = 0; r < 8; ++r) {
                const int row = mBase + i * 16 + r + halfSel * 8;
                float v = acc[i][t][r] + bv;
                if (MODE == 2) v += extra[(size_t)row * N + col];
                if (MODE == 0)
                    ((float*)out)[(size_t)row * N + col] = v;
                else
                    ((__bf16*)out)[(size_t)row * N + col] = (__bf16)v;
            }
        }
    }
}

// ---------------------------------------------------------------------------
// 256-thread block reduction (sum).
// ---------------------------------------------------------------------------
__device__ __forceinline__ float blockReduceSum(float v, float* red) {
    const int tid = threadIdx.x;
    red[tid] = v;
    __syncthreads();
#pragma unroll
    for (int s = 128; s > 0; s >>= 1) {
        if (tid < s) red[tid] += red[tid + s];
        __syncthreads();
    }
    float r = red[0];
    __syncthreads();
    return r;
}

// ---------------------------------------------------------------------------
// Per-row stats: normalize q in place (fp32), k row-norm reciprocal from bf16,
// qw[row] = dot(q_norm, w_g) * scale.  One 256-thread WG per row.
// ---------------------------------------------------------------------------
__global__ __launch_bounds__(256) void eaa_rowstats(float* __restrict__ q,
                                                    const __bf16* __restrict__ kb,
                                                    float* __restrict__ kinv,
                                                    const float* __restrict__ wg,
                                                    float* __restrict__ qw,
                                                    int D, float scale) {
    __shared__ float red[256];
    const int row    = blockIdx.x;
    float* qr        = q + (size_t)row * D;
    const __bf16* kr = kb + (size_t)row * D;

    float sq = 0.f, sk = 0.f;
    for (int d = threadIdx.x; d < D; d += 256) {
        float a = qr[d];
        sq += a * a;
        float b = (float)kr[d];
        sk += b * b;
    }
    float tq = blockReduceSum(sq, red);
    float tk = blockReduceSum(sk, red);
    const float qinv = 1.f / fmaxf(sqrtf(tq), EPS);
    if (threadIdx.x == 0) kinv[row] = 1.f / fmaxf(sqrtf(tk), EPS);

    float dot = 0.f;
    for (int d = threadIdx.x; d < D; d += 256) {
        float a = qr[d] * qinv;
        qr[d]   = a;  // q normalized in place
        dot += a * wg[d];
    }
    float td = blockReduceSum(dot, red);
    if (threadIdx.x == 0) qw[row] = td * scale;
}

// ---------------------------------------------------------------------------
// Per-batch: ainv[b] = 1 / max(||qw[b,:]||_2, eps) over N.
// ---------------------------------------------------------------------------
__global__ __launch_bounds__(256) void eaa_batchnorm(const float* __restrict__ qw,
                                                     float* __restrict__ ainv, int N) {
    __shared__ float red[256];
    const int b = blockIdx.x;
    float s = 0.f;
    for (int n = threadIdx.x; n < N; n += 256) {
        float v = qw[(size_t)b * N + n];
        s += v * v;
    }
    float t = blockReduceSum(s, red);
    if (threadIdx.x == 0) ainv[b] = 1.f / fmaxf(sqrtf(t), EPS);
}

// ---------------------------------------------------------------------------
// G[b,d] = sum_n (qw[b,n]*ainv[b]) * q_norm[b,n,d].  qw staged in LDS.
// grid = (D/256, B)
// ---------------------------------------------------------------------------
__global__ __launch_bounds__(256) void eaa_gdesc(const float* __restrict__ q,
                                                 const float* __restrict__ qw,
                                                 const float* __restrict__ ainv,
                                                 float* __restrict__ G, int N, int D) {
    __shared__ float aw[2048];
    const int b    = blockIdx.y;
    const int d    = blockIdx.x * 256 + threadIdx.x;
    const float ai = ainv[b];
    for (int n = threadIdx.x; n < N; n += 256) aw[n] = qw[(size_t)b * N + n] * ai;
    __syncthreads();
    float acc = 0.f;
    const float* qb = q + (size_t)b * N * D + d;
    for (int n = 0; n < N; ++n) acc += aw[n] * qb[(size_t)n * D];
    G[(size_t)b * D + d] = acc;
}

// ---------------------------------------------------------------------------
// gkb[b,n,d] = bf16( G[b,d] * k[b,n,d] * kinv[b,n] )
// ---------------------------------------------------------------------------
__global__ __launch_bounds__(256) void eaa_scale_k(const __bf16* __restrict__ kb,
                                                   const float* __restrict__ kinv,
                                                   const float* __restrict__ G,
                                                   __bf16* __restrict__ gkb,
                                                   int N, int D, int total) {
    int idx = blockIdx.x * 256 + threadIdx.x;
    if (idx >= total) return;
    int row = idx / D;
    int d   = idx - row * D;
    int b   = row / N;
    gkb[idx] = (__bf16)((float)kb[idx] * kinv[row] * G[(size_t)b * D + d]);
}

// ---------------------------------------------------------------------------
// Launch
// ---------------------------------------------------------------------------
extern "C" void kernel_launch(void* const* d_in, const int* in_sizes, int n_in,
                              void* d_out, int out_size, void* d_ws, size_t ws_size,
                              hipStream_t stream) {
    const float* x   = (const float*)d_in[0];
    const float* Wq  = (const float*)d_in[1];
    const float* bq  = (const float*)d_in[2];
    const float* Wk  = (const float*)d_in[3];
    const float* bk  = (const float*)d_in[4];
    const float* wg  = (const float*)d_in[5];
    const float* Wp  = (const float*)d_in[6];
    const float* bp  = (const float*)d_in[7];
    const float* Wf  = (const float*)d_in[8];
    const float* bfv = (const float*)d_in[9];
    float* out = (float*)d_out;

    constexpr int Bn = 16, Nn = 2048, Dn = 768;
    constexpr int M  = Bn * Nn;           // 32768
    const int total  = M * Dn;            // 25,165,824
    const float scale = 0.036084391824351615f;  // 768^(-1/2)

    // ---- workspace carve-up (256B aligned) ----
    char* ws = (char*)d_ws;
    size_t off = 0;
    auto alloc = [&](size_t bytes) {
        void* p = ws + off;
        off += (bytes + 255) & ~(size_t)255;
        return p;
    };
    float*  qbuf = (float*)alloc((size_t)total * 4);   // q fp32, normalized in place
    __bf16* kbuf = (__bf16*)alloc((size_t)total * 2);  // k bf16 (raw)
    __bf16* xb   = (__bf16*)alloc((size_t)total * 2);  // x bf16
    __bf16* WqT  = (__bf16*)alloc((size_t)Dn * Dn * 2);
    __bf16* WkT  = (__bf16*)alloc((size_t)Dn * Dn * 2);
    __bf16* WpT  = (__bf16*)alloc((size_t)Dn * Dn * 2);
    __bf16* WfT  = (__bf16*)alloc((size_t)Dn * Dn * 2);
    float*  qw   = (float*)alloc((size_t)M * 4);
    float*  kinv = (float*)alloc((size_t)M * 4);
    float*  ainv = (float*)alloc((size_t)Bn * 4);
    float*  G    = (float*)alloc((size_t)Bn * Dn * 4);
    __bf16* gkb  = (__bf16*)alloc((size_t)total * 2);
    __bf16* hb   = xb;  // reuse: xb dead after the q/k GEMMs (same stream order)

    // 1) conversions
    eaa_convert_x<<<(total / 4 + 255) / 256, 256, 0, stream>>>(x, xb, total);
    const int wgrid = (Dn * Dn + 255) / 256;
    eaa_transpose_w<<<wgrid, 256, 0, stream>>>(Wq, WqT, Dn, Dn);
    eaa_transpose_w<<<wgrid, 256, 0, stream>>>(Wk, WkT, Dn, Dn);
    eaa_transpose_w<<<wgrid, 256, 0, stream>>>(Wp, WpT, Dn, Dn);
    eaa_transpose_w<<<wgrid, 256, 0, stream>>>(Wf, WfT, Dn, Dn);

    // 2) q = x@Wq + bq (f32), k = x@Wk + bk (bf16)
    const int gemmGrid = (M / 128) * (Dn / 256);  // 256 * 3 = 768
    eaa_gemm_bf16<0><<<gemmGrid, 256, 0, stream>>>(xb, WqT, bq, nullptr, qbuf, M, Dn, Dn);
    eaa_gemm_bf16<1><<<gemmGrid, 256, 0, stream>>>(xb, WkT, bk, nullptr, kbuf, M, Dn, Dn);

    // 3) row-wise norms, qw
    eaa_rowstats<<<M, 256, 0, stream>>>(qbuf, kbuf, kinv, wg, qw, Dn, scale);

    // 4) per-batch A-normalization, global descriptor G
    eaa_batchnorm<<<Bn, 256, 0, stream>>>(qw, ainv, Nn);
    eaa_gdesc<<<dim3(Dn / 256, Bn), 256, 0, stream>>>(qbuf, qw, ainv, G, Nn, Dn);

    // 5) G * k_norm -> bf16
    eaa_scale_k<<<(total + 255) / 256, 256, 0, stream>>>(kbuf, kinv, G, gkb, Nn, Dn, total);

    // 6) h = gk@Wp + bp + q (bf16), out = h@Wf + bf (f32)
    eaa_gemm_bf16<2><<<gemmGrid, 256, 0, stream>>>(gkb, WpT, bp, qbuf, hb, M, Dn, Dn);
    eaa_gemm_bf16<0><<<gemmGrid, 256, 0, stream>>>(hb, WfT, bfv, nullptr, out, M, Dn, Dn);
}
